// NNConvActor_43439299231749
// MI455X (gfx1250) — matrix operand order repaired and checked by reading, not compiled, not run on hardware
//
#include <hip/hip_runtime.h>
#include <hip/hip_bf16.h>

// ---------------------------------------------------------------------------
// NNConv + actor head for MI455X (gfx1250, wave32, WMMA).
// Dominant GEMM [400k,64]@[64,320] fused with the per-edge contraction so the
// 512MB "w" tensor is never materialized; GEMM runs on v_wmma_f32_16x16x32_bf16.
// Round 2: operands pre-swizzled into per-lane WMMA register order so A loads
// are ds_load_b128 and B loads are coalesced global_load_b128 (no u16 gathers).
// ---------------------------------------------------------------------------

typedef __attribute__((ext_vector_type(16))) __bf16          v16bf;
typedef __attribute__((ext_vector_type(16))) unsigned short  v16us;
typedef __attribute__((ext_vector_type(8)))  float           v8f;

#define N_NODES   50000
#define N_EDGES   400000
#define IN_NODE   16
#define OUT_NODE  20
#define HID_E     64
#define WCOLS     320      // IN_NODE*OUT_NODE
#define N_GRAPHS  128
#define BN_EPS    1e-5f

// ---- workspace layout (float offsets). Needs ~3.9 MB of d_ws. ----
#define AGG_OFF    0          // [N_NODES*20] scatter-add target
#define SSUM_OFF   1000000    // [64] BN column sums
#define SSQ_OFF    1000064    // [64] BN column sum-of-squares
#define GSUM_OFF   1000128    // [128*20] per-graph sums
#define GCNT_OFF   1002688    // [128] per-graph counts
#define ZERO_SPAN  1002816
#define SCALE_OFF  1002816    // [64] folded BN scale
#define SHIFT_OFF  1002880    // [64] folded BN shift (includes b1, mu, beta)
#define W2SW_FOFF  1002944    // ushort region: [20*32*2*16] W2 bf16, WMMA-swizzled

__device__ __forceinline__ unsigned short f2bf(float f) {
    unsigned int u = __float_as_uint(f);
    u += 0x7FFFu + ((u >> 16) & 1u);       // round-to-nearest-even
    return (unsigned short)(u >> 16);
}

// ---------------------------------------------------------------------------
__global__ __launch_bounds__(256) void k0_zero(float* __restrict__ ws) {
    int gid = blockIdx.x * 256 + threadIdx.x;
    if (gid < ZERO_SPAN) ws[gid] = 0.0f;
}

// ---------------------------------------------------------------------------
// K1: column sums / sumsq of h = edge_attr@W1 + b1 (for training-mode BN).
__global__ __launch_bounds__(256) void k1_stats(const float* __restrict__ ea,
                                                const float* __restrict__ W1,
                                                const float* __restrict__ b1,
                                                float* __restrict__ ws) {
    __shared__ float rs[256], rq[256];
    const int tid = threadIdx.x;
    const int k   = tid & 63;
    const int gid = blockIdx.x * 4 + (tid >> 6);
    const int stride = gridDim.x * 4;
    float w1r[10];
#pragma unroll
    for (int i = 0; i < 10; ++i) w1r[i] = W1[i * 64 + k];
    const float bb = b1[k];
    float s = 0.f, q = 0.f;
    for (int e = gid; e < N_EDGES; e += stride) {
        const float* ar = ea + (size_t)e * 10;
        float d = bb;
#pragma unroll
        for (int i = 0; i < 10; ++i) d += ar[i] * w1r[i];
        s += d; q += d * d;
    }
    rs[tid] = s; rq[tid] = q;
    __syncthreads();
    if (tid < 64) {
        float S = rs[tid] + rs[tid + 64] + rs[tid + 128] + rs[tid + 192];
        float Q = rq[tid] + rq[tid + 64] + rq[tid + 128] + rq[tid + 192];
        atomicAdd(&ws[SSUM_OFF + tid], S);
        atomicAdd(&ws[SSQ_OFF  + tid], Q);
    }
}

// ---------------------------------------------------------------------------
// K2: fold BN into scale/shift; pre-swizzle W2 -> bf16 in WMMA B-operand order:
//   w2sw[((nt*32 + lane)*2 + op)*16 + e2] = bf16(W2[K*320 + col])
//   with hi=lane>>4, lo=lane&15, K = op*32 + hi*16 + e2, col = nt*16 + lo.
// In K3 each lane then loads one v16us (32B contiguous) per operand.
__global__ __launch_bounds__(256) void k2_finalize(const float* __restrict__ W2,
                                                   const float* __restrict__ gamma,
                                                   const float* __restrict__ beta,
                                                   const float* __restrict__ b1,
                                                   float* __restrict__ ws,
                                                   unsigned short* __restrict__ w2sw) {
    const int q = blockIdx.x * 256 + threadIdx.x;     // 80*256 == 20480 exactly
    {
        const int e2   = q & 15;
        const int op   = (q >> 4) & 1;
        const int lane = (q >> 5) & 31;
        const int nt   = q >> 10;
        const int hi   = lane >> 4, lo = lane & 15;
        const int K    = op * 32 + hi * 16 + e2;
        const int col  = nt * 16 + lo;
        w2sw[q] = f2bf(W2[K * WCOLS + col]);
    }
    if (blockIdx.x == 0 && threadIdx.x < 64) {
        const int k = threadIdx.x;
        const float inv = 1.0f / (float)N_EDGES;
        float mu  = ws[SSUM_OFF + k] * inv;
        float var = ws[SSQ_OFF + k] * inv - mu * mu;     // biased, as BN uses
        float sc  = gamma[k] * rsqrtf(var + BN_EPS);
        ws[SCALE_OFF + k] = sc;
        ws[SHIFT_OFF + k] = beta[k] + sc * (b1[k] - mu); // h_bn = sc*dot + shift
    }
}

// ---------------------------------------------------------------------------
// K3: main fused kernel. 128 threads = 4 waves, 64 edges per block.
//   phase 1: H[64x64] = relu(BN(edge_attr@W1)) as bf16, stored in LDS directly
//            in WMMA A-operand order: Hp[((wv*32+lane)*2+op)*16 + e2].
//   phase 2: per wave, 16x320 = H_tile @ W2 via bf16 WMMA, in 2 chunks of 160
//            cols (= 8 complete i-groups of OUT_NODE=20); contract with x[src]
//            (+ b2 folded in) and scatter-add msg to agg via f32 atomics.
__global__ __launch_bounds__(128) void k3_msg(const float* __restrict__ x,
                                              const int*   __restrict__ ei,
                                              const float* __restrict__ ea,
                                              const float* __restrict__ W1,
                                              const float* __restrict__ b2,
                                              const float* __restrict__ scale,
                                              const float* __restrict__ shift,
                                              const unsigned short* __restrict__ w2sw,
                                              float* __restrict__ agg) {
    __shared__ unsigned short Hp[64 * 64];     // bf16 H, A-operand swizzled (8KB)
    __shared__ float Ws[4][16 * 160];          // per-wave w-chunk (40KB)
    __shared__ float b2s[WCOLS];
    const int tid = threadIdx.x;
    const int e0  = blockIdx.x * 64;

    __builtin_prefetch(ea + (size_t)(e0 + 64) * 10, 0, 1);   // global_prefetch_b8

    for (int i = tid; i < WCOLS; i += 128) b2s[i] = b2[i];

    { // phase 1: each thread owns one column k for 32 edges; store swizzled.
        const int k  = tid & 63;
        const int eg = tid >> 6;
        // inverse of the A-operand mapping K(e2,hi,op): constant per thread
        const int a_op = k >> 5;                    // operand 0 (K<32) / 1
        const int kk   = k & 31;
        const int grp  = kk >> 4;                   // half-wave
        const int rem  = kk & 15;
        const int hi   = grp;                       // half-wave
        const int e2   = rem;
        const int base = hi * (16 * 32) + a_op * 16 + e2;   // + lo*32 + wv*1024
        float w1r[10];
#pragma unroll
        for (int i = 0; i < 10; ++i) w1r[i] = W1[i * 64 + k];
        const float sc = scale[k], sh = shift[k];
        for (int el = eg; el < 64; el += 2) {
            const float* ar = ea + (size_t)(e0 + el) * 10;
            float d = 0.f;
#pragma unroll
            for (int i = 0; i < 10; ++i) d += ar[i] * w1r[i];
            const int wv2 = el >> 4, lo2 = el & 15;
            Hp[wv2 * 1024 + lo2 * 32 + base] = f2bf(fmaxf(0.f, sc * d + sh));
        }
    }
    __syncthreads();

    const int lane = tid & 31;
    const int wv   = tid >> 5;
    const int lo   = lane & 15;
    const int hi   = lane >> 4;

    // A operands: two contiguous 32B vectors per lane (ds_load_b128 x2 each)
    const v16us* ap = (const v16us*)&Hp[wv * 1024 + lane * 32];
    const v16us a0u = ap[0];
    const v16us a1u = ap[1];
    const v16bf A0 = __builtin_bit_cast(v16bf, a0u);
    const v16bf A1 = __builtin_bit_cast(v16bf, a1u);

    // per-lane edge assignment for contraction: edge m, output half oh
    const int m  = lo;
    const int oh = hi;
    const int eg = e0 + wv * 16 + m;
    const int src = ei[eg];
    const int dst = ei[N_EDGES + eg];
    float xr[16];
    {
        const float4* xp = reinterpret_cast<const float4*>(x + (size_t)src * IN_NODE);
        float4 v0 = xp[0], v1 = xp[1], v2 = xp[2], v3 = xp[3];
        xr[0]=v0.x; xr[1]=v0.y; xr[2]=v0.z; xr[3]=v0.w;
        xr[4]=v1.x; xr[5]=v1.y; xr[6]=v1.z; xr[7]=v1.w;
        xr[8]=v2.x; xr[9]=v2.y; xr[10]=v2.z; xr[11]=v2.w;
        xr[12]=v3.x; xr[13]=v3.y; xr[14]=v3.z; xr[15]=v3.w;
    }
    float msg[10];
#pragma unroll
    for (int j = 0; j < 10; ++j) msg[j] = 0.f;

    const v16us* bp = (const v16us*)w2sw;   // pre-swizzled B operands

#pragma unroll
    for (int ch = 0; ch < 2; ++ch) {
        // WMMA phase: 10 N-tiles of 16 cols, K = 64 in two 32-steps
#pragma unroll
        for (int t = 0; t < 10; ++t) {
            const int nt = ch * 10 + t;
            const int bi = (nt * 32 + lane) * 2;
            const v16us b0u = bp[bi];          // 32B contiguous, coalesced
            const v16us b1u = bp[bi + 1];
            const v16bf B0 = __builtin_bit_cast(v16bf, b0u);
            const v16bf B1 = __builtin_bit_cast(v16bf, b1u);
            v8f c = {};
            c = __builtin_amdgcn_wmma_f32_16x16x32_bf16(false, A0, false, B0,
                                                        (short)0, c, false, false);
            c = __builtin_amdgcn_wmma_f32_16x16x32_bf16(false, A1, false, B1,
                                                        (short)0, c, false, false);
#pragma unroll
            for (int r = 0; r < 8; ++r)        // C/D: lane=col, vgpr r = row (+8 hi)
                Ws[wv][(r + hi * 8) * 160 + t * 16 + lo] = c[r];
        }
        __syncthreads();
        // contraction phase: msg[o] += x_i * (w[m, i*20+o] + b2[i*20+o])
#pragma unroll
        for (int i8 = 0; i8 < 8; ++i8) {
            const int i  = ch * 8 + i8;
            const float xv = xr[i];
#pragma unroll
            for (int j = 0; j < 10; ++j) {
                const int o = oh * 10 + j;
                msg[j] += xv * (Ws[wv][m * 160 + i8 * 20 + o] + b2s[i * 20 + o]);
            }
        }
        __syncthreads();
    }
#pragma unroll
    for (int j = 0; j < 10; ++j)
        atomicAdd(&agg[(size_t)dst * OUT_NODE + oh * 10 + j], msg[j]);
}

// ---------------------------------------------------------------------------
// K4: out = agg + x@root + bias, accumulated straight into per-graph pools.
__global__ __launch_bounds__(256) void k4_node(const float* __restrict__ x,
                                               const int*   __restrict__ batch,
                                               const float* __restrict__ root,
                                               const float* __restrict__ bias,
                                               float* __restrict__ ws) {
    const int gid = blockIdx.x * 256 + threadIdx.x;
    if (gid >= N_NODES * OUT_NODE) return;
    const int n = gid / OUT_NODE, o = gid % OUT_NODE;
    float v = ws[AGG_OFF + gid] + bias[o];
    const float* xr = x + (size_t)n * IN_NODE;
#pragma unroll
    for (int i = 0; i < IN_NODE; ++i) v += xr[i] * root[i * OUT_NODE + o];
    const int g = batch[n];
    atomicAdd(&ws[GSUM_OFF + g * OUT_NODE + o], v);
    if (o == 0) atomicAdd(&ws[GCNT_OFF + g], 1.0f);
}

// ---------------------------------------------------------------------------
// K5: actor MLP per graph: relu(pooled@A1 + ab1) @ A2 + ab2.
__global__ __launch_bounds__(256) void k5_actor(const float* __restrict__ A1,
                                                const float* __restrict__ ab1,
                                                const float* __restrict__ A2,
                                                const float* __restrict__ ab2,
                                                const float* __restrict__ ws,
                                                float* __restrict__ out) {
    __shared__ float pooled[OUT_NODE];
    __shared__ float hs[256];
    const int g = blockIdx.x;
    const int t = threadIdx.x;
    if (t < OUT_NODE) {
        float cnt = fmaxf(ws[GCNT_OFF + g], 1.0f);
        pooled[t] = ws[GSUM_OFF + g * OUT_NODE + t] / cnt;
    }
    __syncthreads();
    float h = ab1[t];
#pragma unroll
    for (int o = 0; o < OUT_NODE; ++o) h += pooled[o] * A1[o * 256 + t];
    hs[t] = fmaxf(h, 0.f);
    __syncthreads();
    if (t < 16) {
        float a = ab2[t];
        for (int j = 0; j < 256; ++j) a += hs[j] * A2[j * 16 + t];
        out[g * 16 + t] = a;
    }
}

// ---------------------------------------------------------------------------
extern "C" void kernel_launch(void* const* d_in, const int* in_sizes, int n_in,
                              void* d_out, int out_size, void* d_ws, size_t ws_size,
                              hipStream_t stream) {
    const float* x     = (const float*)d_in[0];
    const int*   ei    = (const int*)  d_in[1];
    const float* ea    = (const float*)d_in[2];
    const int*   batch = (const int*)  d_in[3];
    const float* W1    = (const float*)d_in[4];
    const float* b1    = (const float*)d_in[5];
    const float* gamma = (const float*)d_in[6];
    const float* beta  = (const float*)d_in[7];
    const float* W2    = (const float*)d_in[8];
    const float* b2    = (const float*)d_in[9];
    const float* root  = (const float*)d_in[10];
    const float* bias  = (const float*)d_in[11];
    const float* A1    = (const float*)d_in[12];
    const float* ab1   = (const float*)d_in[13];
    const float* A2    = (const float*)d_in[14];
    const float* ab2   = (const float*)d_in[15];
    float* ws = (float*)d_ws;
    unsigned short* w2sw = (unsigned short*)(ws + W2SW_FOFF);
    float* out = (float*)d_out;

    k0_zero    <<<(ZERO_SPAN + 255) / 256, 256, 0, stream>>>(ws);
    k1_stats   <<<512, 256, 0, stream>>>(ea, W1, b1, ws);
    k2_finalize<<<80, 256, 0, stream>>>(W2, gamma, beta, b1, ws, w2sw);
    k3_msg     <<<N_EDGES / 64, 128, 0, stream>>>(x, ei, ea, W1, b2,
                                                  ws + SCALE_OFF, ws + SHIFT_OFF,
                                                  w2sw, ws + AGG_OFF);
    k4_node    <<<(N_NODES * OUT_NODE + 255) / 256, 256, 0, stream>>>(x, batch, root, bias, ws);
    k5_actor   <<<N_GRAPHS, 256, 0, stream>>>(A1, ab1, A2, ab2, ws, out);
}
